// _MinGRUBlock_37563783971150
// MI455X (gfx1250) — matrix operand order repaired and checked by reading, not compiled, not run on hardware
//
#include <hip/hip_runtime.h>
#include <hip/hip_bf16.h>

// ---------------------------------------------------------------------------
// minGRU block on gfx1250:
//   1) cvt: x, W fp32 -> f16 (one pass; converted operands are L2-resident)
//   2) GEMM hg = x @ W^T : TDM tensor_load_to_lds (padded tiles) + WMMA f16
//   3) log-space chunked parallel scan over S
//   4) LayerNorm over channels
// Shapes: B=4, S=4096, XD=512, YD=512, N2=1024, M=B*S=16384
// ---------------------------------------------------------------------------

#define BB 4
#define SS 4096
#define XDIM 512
#define YDIM 512
#define N2 1024
#define MROWS (BB * SS)          // 16384
#define CHUNK 64                 // scan chunk length
#define NCHUNK (SS / CHUNK)      // 64 chunks

typedef __attribute__((ext_vector_type(16))) _Float16 v16h;
typedef __attribute__((ext_vector_type(8)))  float    v8f;
typedef __attribute__((ext_vector_type(4)))  unsigned int v4u;
typedef __attribute__((ext_vector_type(4)))  int      v4i;
typedef __attribute__((ext_vector_type(8)))  int      v8i;

union FragU { uint4 q[2]; v16h v; };

__device__ __forceinline__ unsigned int pack2h(float a, float b) {
    union { _Float16 h[2]; unsigned int u; } p;
    p.h[0] = (_Float16)a;
    p.h[1] = (_Float16)b;
    return p.u;
}

// ---------------------- fp32 -> f16 conversion pass ------------------------
__global__ __launch_bounds__(256)
void cvt_f32_to_f16(const float* __restrict__ X, const float* __restrict__ W,
                    _Float16* __restrict__ Xh, _Float16* __restrict__ Wh)
{
    const long long j   = (long long)blockIdx.x * 256 + threadIdx.x;  // 8-elt groups
    const long long NX8 = (long long)MROWS * XDIM / 8;                // 1,048,576
    const float4* src;
    _Float16*     dst;
    long long     g;
    if (j < NX8) { src = (const float4*)X; dst = Xh; g = j; }
    else         { src = (const float4*)W; dst = Wh; g = j - NX8; }
    const float4 a = src[g * 2 + 0];
    const float4 b = src[g * 2 + 1];
    uint4 q;
    q.x = pack2h(a.x, a.y);
    q.y = pack2h(a.z, a.w);
    q.z = pack2h(b.x, b.y);
    q.w = pack2h(b.z, b.w);
    *(uint4*)(dst + g * 8) = q;
}

// ------------------------------ GEMM (TDM + WMMA) --------------------------
// C[m][n] = sum_k Xh[m][k] * Wh[n][k]; workgroup tile 128x128, full K in LDS.
// 4 TDM panels: {A,B} x {K 0..255, K 256..511}; each 128 rows x 256 halves,
// DMA-padded to 528B row stride (bank-conflict-free b128 fragment gathers).
#define TM 128
#define TN 128
#define PANEL_ROW_U 132          // uints per padded row (512B data + 16B pad)
#define PANEL_BYTES (128 * PANEL_ROW_U * 4)   // 67,584 B per panel

__device__ __forceinline__ void make_tdesc(unsigned lds_off,
                                           unsigned long long gaddr,
                                           unsigned rows_total,
                                           v4u& g0, v8i& g1)
{
    // D# group 0: count=1 | lds_addr | global_addr[56:0] | type=2
    g0[0] = 1u;
    g0[1] = lds_off;
    g0[2] = (unsigned)gaddr;
    g0[3] = (unsigned)((gaddr >> 32) & 0x1FFFFFFu) | (2u << 30);
    // D# group 1:
    //  w0: data_size=1(2B) | pad_enable | pad_interval=6(128 DW) | pad_amount=3(4 DW)
    g1[0] = (int)((1u << 16) | (1u << 20) | (6u << 22) | (3u << 25));
    g1[1] = (int)(512u << 16);                  // tensor_dim0 = 512 (lo16)
    g1[2] = (int)((rows_total & 0xFFFFu) << 16);// tensor_dim1 (lo16); dim0 hi = 0
    g1[3] = (int)(256u << 16);                  // tile_dim0 = 256; dim1 hi = 0
    g1[4] = (int)128u;                          // tile_dim1 = 128; tile_dim2 = 0
    g1[5] = (int)512u;                          // tensor_dim0_stride = 512
    g1[6] = 0;
    g1[7] = 0;
}

__device__ __forceinline__ void tdm_load(const v4u& g0, const v8i& g1)
{
    const v4i z4 = (v4i)0;
#if __clang_major__ >= 23
    const v8i z8 = (v8i)0;
    __builtin_amdgcn_tensor_load_to_lds(g0, g1, z4, z4, z8, 0);
#else
    __builtin_amdgcn_tensor_load_to_lds(g0, g1, z4, z4, 0);
#endif
}

__global__ __launch_bounds__(256)
void mingru_gemm_tdm(const _Float16* __restrict__ Xh,
                     const _Float16* __restrict__ Wh,
                     float* __restrict__ HG)
{
    // 4 panels x 67,584 B = 270,336 B of the 320 KB WGP LDS
    __shared__ __align__(16) unsigned int lds[4 * 128 * PANEL_ROW_U];

    const int tid  = threadIdx.x;
    const int lane = tid & 31;
    const int wave = tid >> 5;        // 0..7
    const int wm   = wave >> 1;       // 0..3 -> M offset wm*32
    const int wn   = wave & 1;        // 0..1 -> N offset wn*64
    const int m0   = blockIdx.y * TM;
    const int n0   = blockIdx.x * TN;

    const unsigned lds_base = (unsigned)(size_t)(void*)&lds[0];

    // ---- issue 4 TDM descriptors from wave 0 (EXEC-independent DMA) ----
    if (wave == 0) {
        v4u g0; v8i g1;
        // panel 0: A rows [m0,m0+128), K [0,256)
        make_tdesc(lds_base + 0 * PANEL_BYTES,
                   (unsigned long long)(size_t)(Xh + (size_t)m0 * XDIM),
                   MROWS, g0, g1);
        tdm_load(g0, g1);
        // panel 2: B rows [n0,n0+128), K [0,256)
        make_tdesc(lds_base + 2 * PANEL_BYTES,
                   (unsigned long long)(size_t)(Wh + (size_t)n0 * XDIM),
                   N2, g0, g1);
        tdm_load(g0, g1);
        // panel 1: A, K [256,512)
        make_tdesc(lds_base + 1 * PANEL_BYTES,
                   (unsigned long long)(size_t)(Xh + (size_t)m0 * XDIM + 256),
                   MROWS, g0, g1);
        tdm_load(g0, g1);
        // panel 3: B, K [256,512)
        make_tdesc(lds_base + 3 * PANEL_BYTES,
                   (unsigned long long)(size_t)(Wh + (size_t)n0 * XDIM + 256),
                   N2, g0, g1);
        tdm_load(g0, g1);
        __builtin_amdgcn_s_wait_tensorcnt(2);   // K-lower panels (in-order) done
    }
    __syncthreads();

    v8f acc[2][4];
#pragma unroll
    for (int i = 0; i < 2; ++i)
#pragma unroll
        for (int j = 0; j < 4; ++j)
            acc[i][j] = (v8f)0.0f;

    const int halfSel = lane >> 4;    // 0/1 lane group
    const int l16     = lane & 15;

#pragma unroll
    for (int half = 0; half < 2; ++half) {
        if (half == 1) {
            if (wave == 0) __builtin_amdgcn_s_wait_tensorcnt(0);
            __syncthreads();          // K-upper panels visible to all waves
        }
        const unsigned int* pa = &lds[(0 + half) * 128 * PANEL_ROW_U];
        const unsigned int* pb = &lds[(2 + half) * 128 * PANEL_ROW_U];
#pragma unroll
        for (int ks = 0; ks < 8; ++ks) {        // 8 K-steps of 32 within panel
            FragU af[2];
#pragma unroll
            for (int fm = 0; fm < 2; ++fm) {
                const unsigned int* p =
                    pa + (wm * 32 + fm * 16 + l16) * PANEL_ROW_U + ks * 16;
                af[fm].q[0] = *(const uint4*)(p + halfSel * 4);
                af[fm].q[1] = *(const uint4*)(p + 8 + halfSel * 4);
            }
            FragU bf[4];
#pragma unroll
            for (int fn = 0; fn < 4; ++fn) {
                const unsigned int* p =
                    pb + (wn * 64 + fn * 16 + l16) * PANEL_ROW_U + ks * 16;
                bf[fn].q[0] = *(const uint4*)(p + halfSel * 4);
                bf[fn].q[1] = *(const uint4*)(p + 8 + halfSel * 4);
            }
#pragma unroll
            for (int fm = 0; fm < 2; ++fm)
#pragma unroll
                for (int fn = 0; fn < 4; ++fn)
                    acc[fm][fn] = __builtin_amdgcn_wmma_f32_16x16x32_f16(
                        false, af[fm].v, false, bf[fn].v,
                        (short)0, acc[fm][fn], false, false);
        }
    }

    // ---- epilogue: C layout -> lane l: N = l&15, VGPR r: M = r + (l>>4)*8 ----
#pragma unroll
    for (int fm = 0; fm < 2; ++fm)
#pragma unroll
        for (int fn = 0; fn < 4; ++fn) {
            const int n = n0 + wn * 64 + fn * 16 + l16;
#pragma unroll
            for (int r = 0; r < 8; ++r) {
                const int m = m0 + wm * 32 + fm * 16 + halfSel * 8 + r;
                HG[(size_t)m * N2 + n] = acc[fm][fn][r];
            }
        }
}

// ------------------------------ scan math ----------------------------------
__device__ __forceinline__ float softplusf(float x) {
    return log1pf(__expf(-fabsf(x))) + fmaxf(x, 0.0f);
}
__device__ __forceinline__ float log_g(float x) {
    return (x >= 0.0f) ? __logf(x + 0.5f) : -softplusf(-x);
}
__device__ __forceinline__ float logaddexpf(float a, float b) {
    float m = fmaxf(a, b);
    if (m == -__builtin_inff()) return m;
    return m + log1pf(__expf(fminf(a, b) - m));
}

// Pass A: per-chunk aggregate (logA = sum lc, logB = running logaddexp).
__global__ __launch_bounds__(256)
void scan_chunk_agg(const float* __restrict__ HG,
                    float* __restrict__ Aagg, float* __restrict__ Bagg)
{
    const int tid = blockIdx.x * blockDim.x + threadIdx.x;   // B*NCHUNK*YD
    const int n = tid & (YDIM - 1);
    const int c = (tid >> 9) & (NCHUNK - 1);
    const int b = tid >> 15;
    const float* base = HG + (size_t)(b * SS + c * CHUNK) * N2;
    float A = 0.0f, Bg = -__builtin_inff();
    for (int t = 0; t < CHUNK; ++t) {
        const float hpre = base[(size_t)t * N2 + n];
        const float gate = base[(size_t)t * N2 + YDIM + n];
        const float lc = -softplusf(gate);
        const float lv = -softplusf(-gate) + log_g(hpre);
        Bg = logaddexpf(Bg + lc, lv);
        A += lc;
    }
    const size_t idx = (size_t)(b * NCHUNK + c) * YDIM + n;
    Aagg[idx] = A;
    Bagg[idx] = Bg;
}

// Pass B: exclusive scan over chunk aggregates -> per-chunk carry-in log(h).
__global__ __launch_bounds__(256)
void scan_chunk_prefix(const float* __restrict__ Aagg, const float* __restrict__ Bagg,
                       float* __restrict__ carry)
{
    const int tid = blockIdx.x * blockDim.x + threadIdx.x;   // B*YD
    const int n = tid & (YDIM - 1);
    const int b = tid >> 9;
    float lh = -__builtin_inff();
    for (int c = 0; c < NCHUNK; ++c) {
        const size_t idx = (size_t)(b * NCHUNK + c) * YDIM + n;
        carry[idx] = lh;
        lh = logaddexpf(Aagg[idx] + lh, Bagg[idx]);
    }
}

// Pass C: apply carry, emit h = exp(log h).
__global__ __launch_bounds__(256)
void scan_apply(const float* __restrict__ HG, const float* __restrict__ carry,
                float* __restrict__ H)
{
    const int tid = blockIdx.x * blockDim.x + threadIdx.x;
    const int n = tid & (YDIM - 1);
    const int c = (tid >> 9) & (NCHUNK - 1);
    const int b = tid >> 15;
    const float* base = HG + (size_t)(b * SS + c * CHUNK) * N2;
    float lh = carry[(size_t)(b * NCHUNK + c) * YDIM + n];
    for (int t = 0; t < CHUNK; ++t) {
        const float hpre = base[(size_t)t * N2 + n];
        const float gate = base[(size_t)t * N2 + YDIM + n];
        const float lc = -softplusf(gate);
        const float lv = -softplusf(-gate) + log_g(hpre);
        lh = logaddexpf(lc + lh, lv);
        H[(size_t)(b * SS + c * CHUNK + t) * YDIM + n] = __expf(lh);
    }
}

// LayerNorm over channels: one block per (b,s) row, 256 threads x 2 elems.
__global__ __launch_bounds__(256)
void layernorm_rows(const float* __restrict__ H, const float* __restrict__ gamma,
                    const float* __restrict__ beta, float* __restrict__ Y)
{
    const int row = blockIdx.x;
    const float* hr = H + (size_t)row * YDIM;
    const int t = threadIdx.x;
    const float v0 = hr[t];
    const float v1 = hr[t + 256];
    float s  = v0 + v1;
    float s2 = v0 * v0 + v1 * v1;
#pragma unroll
    for (int off = 16; off > 0; off >>= 1) {
        s  += __shfl_xor(s,  off, 32);
        s2 += __shfl_xor(s2, off, 32);
    }
    __shared__ float ws1[8], ws2[8];
    if ((t & 31) == 0) { ws1[t >> 5] = s; ws2[t >> 5] = s2; }
    __syncthreads();
    float ts = 0.0f, ts2 = 0.0f;
#pragma unroll
    for (int i = 0; i < 8; ++i) { ts += ws1[i]; ts2 += ws2[i]; }
    const float mean = ts * (1.0f / YDIM);
    const float var  = ts2 * (1.0f / YDIM) - mean * mean;
    const float rs   = rsqrtf(var + 1e-5f);
    Y[(size_t)row * YDIM + t]       = (v0 - mean) * rs * gamma[t]       + beta[t];
    Y[(size_t)row * YDIM + t + 256] = (v1 - mean) * rs * gamma[t + 256] + beta[t + 256];
}

// ---------------------------------------------------------------------------
extern "C" void kernel_launch(void* const* d_in, const int* in_sizes, int n_in,
                              void* d_out, int out_size, void* d_ws, size_t ws_size,
                              hipStream_t stream)
{
    const float* x     = (const float*)d_in[0];   // [4,4096,512]
    const float* W     = (const float*)d_in[1];   // [1024,512]
    const float* gamma = (const float*)d_in[2];   // [512]
    const float* beta  = (const float*)d_in[3];   // [512]
    float* y = (float*)d_out;                     // [4,4096,512]

    // workspace layout: hg | h | Aagg | Bagg | carry | xh | wh
    float* hg    = (float*)d_ws;                          // 16,777,216 f
    float* h     = hg + (size_t)MROWS * N2;               //  8,388,608 f
    float* Aagg  = h + (size_t)MROWS * YDIM;              //    131,072 f
    float* Bagg  = Aagg + (size_t)BB * NCHUNK * YDIM;     //    131,072 f
    float* carry = Bagg + (size_t)BB * NCHUNK * YDIM;     //    131,072 f
    _Float16* xh = (_Float16*)(carry + (size_t)BB * NCHUNK * YDIM); // 8,388,608 h
    _Float16* wh = xh + (size_t)MROWS * XDIM;                       //   524,288 h

    // 0) one-shot fp32 -> f16 staging (both operands then L2-resident)
    const int nCvt8 = (MROWS * XDIM + N2 * XDIM) / 8;     // 1,114,112
    cvt_f32_to_f16<<<nCvt8 / 256, 256, 0, stream>>>(x, W, xh, wh);

    // 1) GEMM: hg = x @ W^T  (TDM tile DMA + WMMA f16, fp32 accum)
    mingru_gemm_tdm<<<dim3(N2 / TN, MROWS / TM), 256, 0, stream>>>(xh, wh, hg);

    // 2) chunked log-space scan
    const int nAgg = BB * NCHUNK * YDIM;                  // 131,072
    scan_chunk_agg<<<nAgg / 256, 256, 0, stream>>>(hg, Aagg, Bagg);
    scan_chunk_prefix<<<(BB * YDIM) / 256, 256, 0, stream>>>(Aagg, Bagg, carry);
    scan_apply<<<nAgg / 256, 256, 0, stream>>>(hg, carry, h);

    // 3) LayerNorm + affine
    layernorm_rows<<<MROWS, 256, 0, stream>>>(h, gamma, beta, y);
}
